// vaeVectorQuantizer_81174881895067
// MI455X (gfx1250) — compile-verified, matrix operand order checked by reading, not run on hardware
//
#include <hip/hip_runtime.h>
#include <hip/hip_bf16.h>

// Problem constants (fixed by the reference).
#define N_ROWS 32768   // rows of x
#define DIM    256     // contraction dim
#define KCODES 8192    // codebook size
#define KT     8       // DIM / 32   (k-chunks per WMMA K=32)
#define MT     2048    // N_ROWS / 16
#define NT     512     // KCODES / 16
#define NG     128     // 64-col groups (KCODES / 64)
#define BSLAB  16384   // halves per ng slab: 4 ntiles * KT * 32 lanes * 16

typedef __attribute__((ext_vector_type(16))) __bf16 v16bf;
typedef __attribute__((ext_vector_type(8)))  float  v8f;

// ---- bf16 helpers (bit-level, no reliance on scalar __bf16 arithmetic) ----
__device__ __forceinline__ unsigned short f2bf_bits(float f) {
  unsigned u = __float_as_uint(f);
  unsigned r = u + 0x7fffu + ((u >> 16) & 1u);   // round-to-nearest-even
  return (unsigned short)(r >> 16);
}
__device__ __forceinline__ float bfbits2f(unsigned short h) {
  return __uint_as_float(((unsigned)h) << 16);
}
// Order-preserving map f32 -> u32 (total order incl. negatives).
__device__ __forceinline__ unsigned fmap(float f) {
  unsigned u = __float_as_uint(f);
  return (u & 0x80000000u) ? ~u : (u | 0x80000000u);
}

// Async global->LDS 16-byte transfer (gfx1250, tracked by ASYNCcnt).
// LDS dest offset is the low 32 bits of the generic pointer (aperture form).
__device__ __forceinline__ void issue_async_b128(unsigned lds_off,
                                                 const void* gaddr) {
  asm volatile("global_load_async_to_lds_b128 %0, %1, off"
               :: "v"(lds_off), "v"((unsigned long long)(size_t)gaddr)
               : "memory");
}
__device__ __forceinline__ void wait_async0() {
  asm volatile("s_wait_asynccnt 0x0" ::: "memory");
}

// ---------------------------------------------------------------------------
__global__ void k_init_min(unsigned long long* __restrict__ minarr) {
  int i = blockIdx.x * blockDim.x + threadIdx.x;
  if (i < N_ROWS) minarr[i] = ~0ull;
}

// Split x into bf16 hi/lo, pre-swizzled into the wave32 16-bit A-matrix
// layout (ISA 7.12.2): lanes 0-15 -> M=0..15 holding K=0..7,16..23;
// lanes 16-31 -> same M, K=8..15,24..31. One thread = one lane's 16 elems.
__global__ void k_prep_x(const float* __restrict__ x,
                         unsigned short* __restrict__ Ahi,
                         unsigned short* __restrict__ Alo) {
  int t    = blockIdx.x * blockDim.x + threadIdx.x;  // [0, MT*KT*32)
  int lane = t & 31;
  int kt   = (t >> 5) & (KT - 1);
  int mt   = t >> 8;
  if (mt >= MT) return;
  int m  = mt * 16 + (lane & 15);
  int kb = kt * 32 + ((lane >> 4) ? 8 : 0);
  const float* xr = x + (size_t)m * DIM;
  size_t base = ((size_t)(mt * KT + kt) * 32 + lane) * 16;
#pragma unroll
  for (int j = 0; j < 8; ++j) {
    int kk = kb + ((j < 4) ? 2 * j : 16 + 2 * (j - 4));
#pragma unroll
    for (int p = 0; p < 2; ++p) {
      float f = xr[kk + p];
      unsigned short hb = f2bf_bits(f);
      Ahi[base + 2 * j + p] = hb;
      Alo[base + 2 * j + p] = f2bf_bits(f - bfbits2f(hb));
    }
  }
}

// Same split+swizzle for E (B-matrix layout: lane&15 selects the N column).
__global__ void k_prep_e(const float* __restrict__ E,
                         unsigned short* __restrict__ Bhi,
                         unsigned short* __restrict__ Blo) {
  int t    = blockIdx.x * blockDim.x + threadIdx.x;  // [0, NT*KT*32)
  int lane = t & 31;
  int kt   = (t >> 5) & (KT - 1);
  int nt   = t >> 8;
  if (nt >= NT) return;
  int n  = nt * 16 + (lane & 15);
  int kb = kt * 32 + ((lane >> 4) ? 8 : 0);
  size_t base = ((size_t)(nt * KT + kt) * 32 + lane) * 16;
#pragma unroll
  for (int j = 0; j < 8; ++j) {
    int kk = kb + ((j < 4) ? 2 * j : 16 + 2 * (j - 4));
#pragma unroll
    for (int p = 0; p < 2; ++p) {
      float f = E[(size_t)(kk + p) * KCODES + n];
      unsigned short hb = f2bf_bits(f);
      Bhi[base + 2 * j + p] = hb;
      Blo[base + 2 * j + p] = f2bf_bits(f - bfbits2f(hb));
    }
  }
}

// enorm[k] = sum_d E[d,k]^2  (exact f32; coalesced across k).
__global__ void k_enorm(const float* __restrict__ E, float* __restrict__ en) {
  int k = blockIdx.x * blockDim.x + threadIdx.x;
  if (k >= KCODES) return;
  float s = 0.f;
  for (int d = 0; d < DIM; ++d) {
    float v = E[(size_t)d * KCODES + k];
    s = fmaf(v, v, s);
  }
  en[k] = s;
}

// ET[k,d] = E[d,k]  (reads coalesced; makes the final gather coalesced).
__global__ void k_et(const float* __restrict__ E, float* __restrict__ ET) {
  int t = blockIdx.x * blockDim.x + threadIdx.x;  // [0, KCODES*DIM)
  int k = t & (KCODES - 1);
  int d = t >> 13;
  if (d >= DIM) return;
  ET[(size_t)k * DIM + d] = E[(size_t)d * KCODES + k];
}

// ---------------------------------------------------------------------------
// Issue the async copy of one 64-col B slab (hi 32KB + lo 32KB) into LDS.
// 256 threads x 16 b128 transfers.
__device__ __forceinline__ void copy_ng_to_lds(
    const unsigned short* Bhi, const unsigned short* Blo,
    unsigned lbase, int buf, int ng, int t) {
  const char* gh = (const char*)(Bhi + (size_t)ng * BSLAB);
  const char* gl = (const char*)(Blo + (size_t)ng * BSLAB);
  unsigned lh = lbase + (unsigned)buf * 65536u + (unsigned)t * 16u;
  unsigned ll = lh + 32768u;
#pragma unroll
  for (int i = 0; i < 8; ++i) {
    issue_async_b128(lh + (unsigned)i * 4096u, gh + (size_t)t * 16 + (size_t)i * 4096);
    issue_async_b128(ll + (unsigned)i * 4096u, gl + (size_t)t * 16 + (size_t)i * 4096);
  }
}

// Persistent fused bf16x3 GEMM + argmin. One block = 8 waves = 128 rows,
// sweeping all 128 col-groups. A panel (16x256 hi+lo) lives in VGPRs for the
// whole kernel; B slabs are double-buffered through LDS via async-to-LDS DMA.
__global__ void __launch_bounds__(256, 1) k_gemm(
    const unsigned short* __restrict__ Ahi,
    const unsigned short* __restrict__ Alo,
    const unsigned short* __restrict__ Bhi,
    const unsigned short* __restrict__ Blo,
    const float* __restrict__ enorm,
    unsigned long long* __restrict__ minarr) {
  __shared__ __align__(64) unsigned short ldsB[2][2][BSLAB];  // 128 KB

  int t    = threadIdx.x;
  int lane = t & 31;
  int wid  = t >> 5;
  int mt   = blockIdx.x * 8 + wid;  // this wave's 16-row tile

  // Whole A panel in registers (read from memory exactly once).
  v16bf ah[KT], al[KT];
#pragma unroll
  for (int kt = 0; kt < KT; ++kt) {
    size_t abase = ((size_t)(mt * KT + kt) * 32 + lane) * 16;
    ah[kt] = *(const v16bf*)(Ahi + abase);
    al[kt] = *(const v16bf*)(Alo + abase);
  }

  unsigned lbase = (unsigned)(size_t)(void*)&ldsB[0][0][0];
  copy_ng_to_lds(Bhi, Blo, lbase, 0, 0, t);
  wait_async0();
  __syncthreads();

  int rowbase = mt * 16 + 8 * (lane >> 4);

  for (int ng = 0; ng < NG; ++ng) {
    int cur = ng & 1;
    if (ng + 1 < NG) copy_ng_to_lds(Bhi, Blo, lbase, cur ^ 1, ng + 1, t);

    const unsigned short* bplane = &ldsB[cur][0][0];
    v8f c[4] = {};
#pragma unroll
    for (int kt = 0; kt < KT; ++kt) {
#pragma unroll
      for (int j = 0; j < 4; ++j) {
        const unsigned short* bp =
            bplane + ((size_t)(j * KT + kt) * 32 + lane) * 16;
        v16bf bh = *(const v16bf*)bp;
        v16bf bl = *(const v16bf*)(bp + BSLAB);  // lo plane of same buffer
        c[j] = __builtin_amdgcn_wmma_f32_16x16x32_bf16(
            false, ah[kt], false, bh, (short)0, c[j], false, false);
        c[j] = __builtin_amdgcn_wmma_f32_16x16x32_bf16(
            false, ah[kt], false, bl, (short)0, c[j], false, false);
        c[j] = __builtin_amdgcn_wmma_f32_16x16x32_bf16(
            false, al[kt], false, bh, (short)0, c[j], false, false);
      }
    }

    // Fused argmin epilogue.
    // C layout: VGPR r, lane l -> row = r + 8*(l/16), col = l%16 per n-tile.
    int cb = ng * 64 + (lane & 15);
    float en0 = enorm[cb];
    float en1 = enorm[cb + 16];
    float en2 = enorm[cb + 32];
    float en3 = enorm[cb + 48];
#pragma unroll
    for (int r = 0; r < 8; ++r) {
      float d0 = en0 - 2.f * c[0][r];
      unsigned long long best =
          ((unsigned long long)fmap(d0) << 32) | (unsigned)cb;
      {
        float d = en1 - 2.f * c[1][r];
        unsigned long long key =
            ((unsigned long long)fmap(d) << 32) | (unsigned)(cb + 16);
        if (key < best) best = key;
      }
      {
        float d = en2 - 2.f * c[2][r];
        unsigned long long key =
            ((unsigned long long)fmap(d) << 32) | (unsigned)(cb + 32);
        if (key < best) best = key;
      }
      {
        float d = en3 - 2.f * c[3][r];
        unsigned long long key =
            ((unsigned long long)fmap(d) << 32) | (unsigned)(cb + 48);
        if (key < best) best = key;
      }
#pragma unroll
      for (int s = 1; s < 16; s <<= 1) {
        unsigned long long o = __shfl_xor(best, s, 32);
        if (o < best) best = o;
      }
      if ((lane & 15) == 0) atomicMin(&minarr[rowbase + r], best);
    }

    wait_async0();   // next buffer's DMA complete (this wave's issues)
    __syncthreads(); // all waves' DMA visible; safe to flip buffers
  }
}

// Coalesced gather of the winning codebook row.
__global__ void k_gather(const float* __restrict__ ET,
                         const unsigned long long* __restrict__ minarr,
                         float* __restrict__ out) {
  int n = blockIdx.x;
  int d = threadIdx.x;
  unsigned idx = (unsigned)(minarr[n] & 0xffffffffu);
  out[(size_t)n * DIM + d] = ET[(size_t)idx * DIM + d];
}

// ---------------------------------------------------------------------------
extern "C" void kernel_launch(void* const* d_in, const int* in_sizes, int n_in,
                              void* d_out, int out_size, void* d_ws,
                              size_t ws_size, hipStream_t stream) {
  const float* x = (const float*)d_in[0];  // [N, D]
  const float* E = (const float*)d_in[1];  // [D, K]
  float* out = (float*)d_out;              // [N, D]

  unsigned char* w = (unsigned char*)d_ws;
  size_t off = 0;
  unsigned short* Ahi = (unsigned short*)(w + off); off += (size_t)N_ROWS * DIM * 2;
  unsigned short* Alo = (unsigned short*)(w + off); off += (size_t)N_ROWS * DIM * 2;
  unsigned short* Bhi = (unsigned short*)(w + off); off += (size_t)KCODES * DIM * 2;
  unsigned short* Blo = (unsigned short*)(w + off); off += (size_t)KCODES * DIM * 2;
  float* ET    = (float*)(w + off);                 off += (size_t)KCODES * DIM * 4;
  float* enorm = (float*)(w + off);                 off += (size_t)KCODES * 4;
  unsigned long long* minarr = (unsigned long long*)(w + off);
  off += (size_t)N_ROWS * 8;  // total ~48.3 MB

  k_init_min<<<N_ROWS / 256, 256, 0, stream>>>(minarr);
  k_prep_x<<<(MT * KT * 32) / 256, 256, 0, stream>>>(x, Ahi, Alo);
  k_prep_e<<<(NT * KT * 32) / 256, 256, 0, stream>>>(E, Bhi, Blo);
  k_enorm<<<KCODES / 256, 256, 0, stream>>>(E, enorm);
  k_et<<<(KCODES * DIM) / 256, 256, 0, stream>>>(E, ET);

  k_gemm<<<MT / 8, 256, 0, stream>>>(Ahi, Alo, Bhi, Blo, enorm, minarr);

  k_gather<<<N_ROWS, 256, 0, stream>>>(ET, minarr, out);
}
